// MultiHeadAttentionSimplicial_49417893707891
// MI455X (gfx1250) — matrix operand order repaired and checked by reading, not compiled, not run on hardware
//
#include <hip/hip_runtime.h>
#include <hip/hip_bf16.h>

// MI455X / gfx1250 implementation of the simplicial multi-head attention.
// Pipeline (all f32 inputs):
//   k_proj : Q,K1,K2,V1,V2 = proj(x/y/z) + bias, head-major [5][32][128][64]   (v_wmma_f32_16x16x4_f32)
//   k_norms: qq, k1k1, k2k2 row norms
//   k_gram : QK1, QK2, K1K2 [32][128][128]                                     (v_wmma_f32_16x16x4_f32)
//   k_bv   : Bvjvk[a][q][j*128+k] in bf16 via LDS-staged Bvj                   (v_wmma_f32_16x16x4_f32)
//   k_attn : two-pass softmax over cubic logits fused with
//            out[i,q] = (1/Z_i) * P[i,(jk)] @ Bvjvk[(jk),q]                    (v_wmma_f32_16x16x32_bf16)

typedef float  v2f   __attribute__((ext_vector_type(2)));
typedef float  v8f   __attribute__((ext_vector_type(8)));
typedef __bf16 v16bf __attribute__((ext_vector_type(16)));
typedef __bf16 v8bf  __attribute__((ext_vector_type(8)));

constexpr int kB  = 4;
constexpr int kS  = 128;
constexpr int kD  = 512;
constexpr int kH  = 8;
constexpr int kA  = 32;   // B*H
constexpr int kDH = 64;

static __device__ __forceinline__ v8f wmma_f32x4(v2f a, v2f b, v8f c) {
  return __builtin_amdgcn_wmma_f32_16x16x4_f32(false, a, false, b, (short)0, c, false, false);
}
static __device__ __forceinline__ v8f wmma_bf16x32(v16bf a, v16bf b, v8f c) {
  return __builtin_amdgcn_wmma_f32_16x16x32_bf16(false, a, false, b, (short)0, c, false, false);
}

// --------------------------------------------------------------------------
// Projections: out[t][a][s][dh] = (src @ W)[row][h*64+dh] + bias
// Block = 128 threads (4 waves); one block per (t, a, s-tile16); each wave one
// 16x16 N-tile, K=512 via f32 WMMA.
// --------------------------------------------------------------------------
__global__ void __launch_bounds__(128)
k_proj(const float* __restrict__ x, const float* __restrict__ y, const float* __restrict__ z,
       const float* __restrict__ Wq,  const float* __restrict__ bq,
       const float* __restrict__ Wk1, const float* __restrict__ bk1,
       const float* __restrict__ Wk2, const float* __restrict__ bk2,
       const float* __restrict__ Wv1, const float* __restrict__ bv1,
       const float* __restrict__ Wv2, const float* __restrict__ bv2,
       float* __restrict__ proj) {
  int bid = blockIdx.x;
  int t  = bid / (kA * 8);
  int a  = (bid / 8) % kA;
  int st = bid % 8;
  int b = a / kH, h = a % kH;
  int lane = threadIdx.x & 31;
  int wave = threadIdx.x >> 5;
  int n = lane & 15;
  int koff = (lane >= 16) ? 2 : 0;

  const float* src; const float* W; const float* bias;
  switch (t) {
    case 0:  src = x; W = Wq;  bias = bq;  break;
    case 1:  src = y; W = Wk1; bias = bk1; break;
    case 2:  src = z; W = Wk2; bias = bk2; break;
    case 3:  src = y; W = Wv1; bias = bv1; break;
    default: src = z; W = Wv2; bias = bv2; break;
  }
  int row = b * kS + st * 16 + n;       // A-matrix row (M = lane&15)
  int col = h * kDH + wave * 16 + n;    // B-matrix col (N = lane&15)
  const float* arow = src + (size_t)row * kD;
  v8f c = {};
  for (int kk = 0; kk < kD; kk += 4) {
    v2f av = *(const v2f*)(arow + kk + koff);
    v2f bv;
    bv.x = W[(size_t)(kk + koff) * kD + col];
    bv.y = W[(size_t)(kk + koff + 1) * kD + col];
    c = wmma_f32x4(av, bv, c);
  }
  float bb = bias[col];
  float* outp = proj + ((size_t)t * kA + a) * (kS * kDH);
  int srow0 = st * 16 + ((lane >= 16) ? 8 : 0);
  int ocol  = wave * 16 + n;
  for (int p = 0; p < 8; ++p)
    outp[(size_t)(srow0 + p) * kDH + ocol] = c[p] + bb;
}

// --------------------------------------------------------------------------
// Row squared norms: nrm[t][a][r], t: 0=Q, 1=K1, 2=K2
// --------------------------------------------------------------------------
__global__ void k_norms(const float* __restrict__ proj, float* __restrict__ nrm) {
  int a = blockIdx.x;
  int t = threadIdx.x / kS;
  int r = threadIdx.x % kS;
  const float* rowp = proj + ((size_t)t * kA + a) * (kS * kDH) + (size_t)r * kDH;
  float s = 0.f;
  for (int d = 0; d < kDH; ++d) { float v = rowp[d]; s = fmaf(v, v, s); }
  nrm[(size_t)t * (kA * kS) + a * kS + r] = s;
}

// --------------------------------------------------------------------------
// Gram matrices: p=0: QK1 = Q@K1^T, p=1: QK2 = Q@K2^T, p=2: K1K2 = K1@K2^T
// Block = 128 threads; each wave one 16x16 tile, K=64 f32 WMMA.
// --------------------------------------------------------------------------
__global__ void __launch_bounds__(128)
k_gram(const float* __restrict__ proj, float* __restrict__ grm) {
  int bid = blockIdx.x;
  int p  = bid / (kA * 16);
  int a  = (bid / 16) % kA;
  int tb = bid % 16;
  int mt = tb >> 1;
  int ng = tb & 1;
  int lane = threadIdx.x & 31;
  int wave = threadIdx.x >> 5;
  int nt = ng * 4 + wave;
  int n = lane & 15;
  int koff = (lane >= 16) ? 2 : 0;
  int lt = (p == 2) ? 1 : 0;
  int rt = (p == 0) ? 1 : 2;
  const float* L = proj + ((size_t)lt * kA + a) * (kS * kDH);
  const float* R = proj + ((size_t)rt * kA + a) * (kS * kDH);
  const float* arow = L + (size_t)(mt * 16 + n) * kDH;
  const float* brow = R + (size_t)(nt * 16 + n) * kDH;   // B[K][N] = R[j=N][K] -> contiguous
  v8f c = {};
  for (int kk = 0; kk < kDH; kk += 4) {
    v2f av = *(const v2f*)(arow + kk + koff);
    v2f bv = *(const v2f*)(brow + kk + koff);
    c = wmma_f32x4(av, bv, c);
  }
  float* outp = grm + ((size_t)p * kA + a) * (kS * kS);
  int r0 = mt * 16 + ((lane >= 16) ? 8 : 0);
  for (int q = 0; q < 8; ++q)
    outp[(size_t)(r0 + q) * kS + nt * 16 + n] = c[q];
}

// --------------------------------------------------------------------------
// Bvjvk[a][q][j*128+k] (bf16) = sum_s (sum_r v1[j,r] * Bw[q,r,s]) * v2[k,s]
// Per (a,q) block of 256 threads (8 waves): phase-1 stages Bvj[j][s] in LDS.
// --------------------------------------------------------------------------
__global__ void __launch_bounds__(256)
k_bv(const float* __restrict__ proj, const float* __restrict__ Bw,
     __bf16* __restrict__ bvjvk) {
  __shared__ float lds_bvj[kS * kDH];   // 32KB  [j][s]
  int a = blockIdx.x / kDH;
  int q = blockIdx.x % kDH;
  int lane = threadIdx.x & 31;
  int wave = threadIdx.x >> 5;
  int n = lane & 15;
  int koff = (lane >= 16) ? 2 : 0;
  const float* V1 = proj + ((size_t)3 * kA + a) * (kS * kDH);
  const float* V2 = proj + ((size_t)4 * kA + a) * (kS * kDH);
  const float* Bq = Bw + (size_t)q * (kDH * kDH);

  { // phase 1: Bvj[j][s] = v1 @ Bw[q]   (M=j, K=r, N=s)
    int jt = wave;
    const float* arow = V1 + (size_t)(jt * 16 + n) * kDH;
    for (int stt = 0; stt < 4; ++stt) {
      int s = stt * 16 + n;
      v8f c = {};
      for (int kk = 0; kk < kDH; kk += 4) {
        v2f av = *(const v2f*)(arow + kk + koff);
        v2f bv;
        bv.x = Bq[(size_t)(kk + koff) * kDH + s];
        bv.y = Bq[(size_t)(kk + koff + 1) * kDH + s];
        c = wmma_f32x4(av, bv, c);
      }
      int r0 = jt * 16 + ((lane >= 16) ? 8 : 0);
      for (int p = 0; p < 8; ++p)
        lds_bvj[(r0 + p) * kDH + s] = c[p];
    }
  }
  __syncthreads();
  { // phase 2: Bvjvk[j][k] = Bvj @ v2^T   (M=j, K=s, N=k; B[K][N]=v2[k][s] contiguous)
    int jt = wave;
    const float* arow = lds_bvj + (size_t)(jt * 16 + n) * kDH;
    __bf16* outp = bvjvk + ((size_t)a * kDH + q) * (kS * kS);
    for (int kt = 0; kt < 8; ++kt) {
      const float* brow = V2 + (size_t)(kt * 16 + n) * kDH;
      v8f c = {};
      for (int kk = 0; kk < kDH; kk += 4) {
        v2f av = *(const v2f*)(arow + kk + koff);
        v2f bv = *(const v2f*)(brow + kk + koff);
        c = wmma_f32x4(av, bv, c);
      }
      int r0 = jt * 16 + ((lane >= 16) ? 8 : 0);
      for (int p = 0; p < 8; ++p)
        outp[(size_t)(r0 + p) * kS + kt * 16 + n] = (__bf16)c[p];
    }
  }
}

// --------------------------------------------------------------------------
// Fused logits + joint softmax + contraction with Bvjvk (bf16 WMMA).
// Block = 256 threads (8 waves) per (a, i-tile of 16 rows).
// --------------------------------------------------------------------------
static __device__ __forceinline__ float logit_f(float qk1, float qk2, float k1k2,
                                                float qq, float k11, float k22) {
  float pre = qk1 * qk1 * k22 + k1k2 * k1k2 * qq + qk2 * qk2 * k11
              - 2.0f * qk1 * qk2 * k1k2;
  if (__builtin_isinf(pre)) pre = 55500.0f;
  pre = fmaxf(pre, 0.0f);
  return __builtin_sqrtf(pre);
}

__global__ void __launch_bounds__(256)
k_attn(const float* __restrict__ grm, const float* __restrict__ nrm,
       const __bf16* __restrict__ bvjvk, float* __restrict__ out) {
  __shared__ float  l_qk1[16 * kS];       // 8KB
  __shared__ float  l_qk2[16 * kS];       // 8KB
  __shared__ float  l_k1k2[16 * kS];      // 8KB (j-tile x k)
  __shared__ float  l_k1k1[kS];
  __shared__ float  l_k2k2[kS];
  __shared__ float  l_qq[16];
  __shared__ float  l_m[16];
  __shared__ float  l_z[16];
  __shared__ __bf16 l_P[16 * 2048];       // 64KB P-chunk (i x (16 j * 128 k))
  __shared__ float  l_red[8 * 16 * 64];   // 32KB cross-wave C reduction

  int a  = blockIdx.x >> 3;
  int it = blockIdx.x & 7;
  int b = a / kH, h = a % kH;
  int tid = threadIdx.x;
  int lane = tid & 31;
  int wave = tid >> 5;
  int n = lane & 15;
  int hi = (lane >= 16);

  const float* QK1  = grm + ((size_t)0 * kA + a) * (kS * kS);
  const float* QK2  = grm + ((size_t)1 * kA + a) * (kS * kS);
  const float* K1K2 = grm + ((size_t)2 * kA + a) * (kS * kS);
  const float* qqp   = nrm + 0 * (kA * kS) + a * kS;
  const float* k1k1p = nrm + 1 * (kA * kS) + a * kS;
  const float* k2k2p = nrm + 2 * (kA * kS) + a * kS;

  for (int idx = tid; idx < 16 * kS; idx += 256) {
    int i = idx >> 7, j = idx & 127;
    l_qk1[idx] = QK1[(size_t)(it * 16 + i) * kS + j];
    l_qk2[idx] = QK2[(size_t)(it * 16 + i) * kS + j];
  }
  for (int idx = tid; idx < kS; idx += 256) {
    l_k1k1[idx] = k1k1p[idx];
    l_k2k2[idx] = k2k2p[idx];
  }
  if (tid < 16) l_qq[tid] = qqp[it * 16 + tid];
  __syncthreads();

  int iloc0 = 2 * wave, iloc1 = 2 * wave + 1;   // two i-rows per wave
  float qq0 = l_qq[iloc0], qq1 = l_qq[iloc1];

  // ---- pass A: row maxima over the 128x128 (j,k) plane ----
  float mx0 = -1e30f, mx1 = -1e30f;
  for (int jt = 0; jt < 8; ++jt) {
    for (int idx = tid; idx < 16 * kS; idx += 256) {
      int jj = idx >> 7, k = idx & 127;
      l_k1k2[idx] = K1K2[(size_t)(jt * 16 + jj) * kS + k];
    }
    __syncthreads();
    for (int jj = 0; jj < 16; ++jj) {
      int j = jt * 16 + jj;
      float a0 = l_qk1[iloc0 * kS + j], a1 = l_qk1[iloc1 * kS + j];
      float k11 = l_k1k1[j];
      for (int c = 0; c < 4; ++c) {
        int k = c * 32 + lane;
        float kk2 = l_k1k2[jj * kS + k];
        float k22 = l_k2k2[k];
        mx0 = fmaxf(mx0, logit_f(a0, l_qk2[iloc0 * kS + k], kk2, qq0, k11, k22));
        mx1 = fmaxf(mx1, logit_f(a1, l_qk2[iloc1 * kS + k], kk2, qq1, k11, k22));
      }
    }
    __syncthreads();
  }
  for (int off = 16; off > 0; off >>= 1) {
    mx0 = fmaxf(mx0, __shfl_xor(mx0, off, 32));
    mx1 = fmaxf(mx1, __shfl_xor(mx1, off, 32));
  }
  if (lane == 0) { l_m[iloc0] = mx0; l_m[iloc1] = mx1; }
  __syncthreads();

  // ---- pass B: P = exp(logit - m), staged to LDS in bf16, fused GEMM ----
  float m0 = l_m[iloc0], m1 = l_m[iloc1];
  float z0 = 0.f, z1 = 0.f;
  v8f cacc[4] = {};
  const __bf16* Bbase = bvjvk + (size_t)a * (kDH * kS * kS);

  for (int jt = 0; jt < 8; ++jt) {
    for (int idx = tid; idx < 16 * kS; idx += 256) {
      int jj = idx >> 7, k = idx & 127;
      l_k1k2[idx] = K1K2[(size_t)(jt * 16 + jj) * kS + k];
    }
    __syncthreads();
    for (int jj = 0; jj < 16; ++jj) {
      int j = jt * 16 + jj;
      float a0 = l_qk1[iloc0 * kS + j], a1 = l_qk1[iloc1 * kS + j];
      float k11 = l_k1k1[j];
      for (int c = 0; c < 4; ++c) {
        int k = c * 32 + lane;
        float kk2 = l_k1k2[jj * kS + k];
        float k22 = l_k2k2[k];
        float e0 = __expf(logit_f(a0, l_qk2[iloc0 * kS + k], kk2, qq0, k11, k22) - m0);
        float e1 = __expf(logit_f(a1, l_qk2[iloc1 * kS + k], kk2, qq1, k11, k22) - m1);
        z0 += e0; z1 += e1;
        l_P[iloc0 * 2048 + jj * kS + k] = (__bf16)e0;
        l_P[iloc1 * 2048 + jj * kS + k] = (__bf16)e1;
      }
    }
    __syncthreads();
    // GEMM over this chunk: wave handles K sub-range [wave*256, wave*256+256)
    if (jt < 7)
      __builtin_prefetch(Bbase + (size_t)n * (kS * kS) + (jt + 1) * 2048 + wave * 256, 0, 1);
    for (int s32 = 0; s32 < 8; ++s32) {
      int kk0 = wave * 256 + s32 * 32;
      union { v16bf v; v8bf half2[2]; } av;
      const __bf16* prow = l_P + n * 2048 + kk0 + (hi ? 16 : 0);
      av.half2[0] = *(const v8bf*)(prow);
      av.half2[1] = *(const v8bf*)(prow + 8);
      for (int qtt = 0; qtt < 4; ++qtt) {
        int qcol = qtt * 16 + n;
        const __bf16* bp = Bbase + (size_t)qcol * (kS * kS) + jt * 2048 + kk0 + (hi ? 16 : 0);
        union { v16bf v; v8bf half2[2]; } bv_;
        bv_.half2[0] = *(const v8bf*)(bp);
        bv_.half2[1] = *(const v8bf*)(bp + 8);
        cacc[qtt] = wmma_bf16x32(av.v, bv_.v, cacc[qtt]);
      }
    }
    __syncthreads();
  }

  // ---- reductions and output ----
  for (int off = 16; off > 0; off >>= 1) {
    z0 += __shfl_xor(z0, off, 32);
    z1 += __shfl_xor(z1, off, 32);
  }
  if (lane == 0) { l_z[iloc0] = z0; l_z[iloc1] = z1; }
  for (int qtt = 0; qtt < 4; ++qtt) {
    int r0 = hi ? 8 : 0;
    for (int p = 0; p < 8; ++p)
      l_red[(wave * 16 + r0 + p) * kDH + qtt * 16 + n] = cacc[qtt][p];
  }
  __syncthreads();
  for (int idx = tid; idx < 16 * kDH; idx += 256) {
    int r = idx >> 6, cc = idx & 63;
    float s = 0.f;
    for (int w = 0; w < 8; ++w) s += l_red[(w * 16 + r) * kDH + cc];
    int srow = it * 16 + r;
    out[((size_t)(b * kS + srow)) * kD + h * kDH + cc] = s / l_z[r];
  }
}

// --------------------------------------------------------------------------
extern "C" void kernel_launch(void* const* d_in, const int* in_sizes, int n_in,
                              void* d_out, int out_size, void* d_ws, size_t ws_size,
                              hipStream_t stream) {
  const float* x   = (const float*)d_in[0];
  const float* y   = (const float*)d_in[1];
  const float* z   = (const float*)d_in[2];
  const float* Bw  = (const float*)d_in[3];
  const float* Wq  = (const float*)d_in[4];  const float* bq  = (const float*)d_in[5];
  const float* Wk1 = (const float*)d_in[6];  const float* bk1 = (const float*)d_in[7];
  const float* Wk2 = (const float*)d_in[8];  const float* bk2 = (const float*)d_in[9];
  const float* Wv1 = (const float*)d_in[10]; const float* bv1 = (const float*)d_in[11];
  const float* Wv2 = (const float*)d_in[12]; const float* bv2 = (const float*)d_in[13];
  float* out = (float*)d_out;

  // Workspace layout (aligned: float region size is divisible by 16 bytes)
  float* proj = (float*)d_ws;                                // 5*32*128*64 f32   (5 MB)
  float* grm  = proj + (size_t)5 * kA * kS * kDH;            // 3*32*128*128 f32  (6 MB)
  float* nrm  = grm  + (size_t)3 * kA * kS * kS;             // 3*32*128 f32
  __bf16* bvjvk = (__bf16*)(nrm + (size_t)3 * kA * kS);      // 32*64*128*128 bf16 (64 MB)

  k_proj <<<5 * kA * 8, 128, 0, stream>>>(x, y, z, Wq, bq, Wk1, bk1, Wk2, bk2,
                                          Wv1, bv1, Wv2, bv2, proj);
  k_norms<<<kA, 3 * kS, 0, stream>>>(proj, nrm);
  k_gram <<<3 * kA * 16, 128, 0, stream>>>(proj, grm);
  k_bv   <<<kA * kDH, 256, 0, stream>>>(proj, Bw, bvjvk);
  k_attn <<<kA * 8, 256, 0, stream>>>(grm, nrm, bvjvk, out);
}
